// MultiHeadAttention_60455959659083
// MI455X (gfx1250) — matrix operand order repaired and checked by reading, not compiled, not run on hardware
//
#include <hip/hip_runtime.h>
#include <math.h>

// ---------------------------------------------------------------------------
// MI455X (gfx1250) multi-head attention, bf16 WMMA pipeline, wave32.
// ---------------------------------------------------------------------------

typedef __attribute__((ext_vector_type(16))) __bf16 v16bf;
typedef __attribute__((ext_vector_type(8)))  __bf16 v8bf;
typedef __attribute__((ext_vector_type(8)))  float  v8f;

#define D_MODEL  1024
#define N_HEADS  16
#define HEAD_DIM 64
#define BATCH    4
#define SEQ      2048
#define M_TOT    (BATCH * SEQ)   // 8192 rows

// ---------------------------------------------------------------------------
// helpers
// ---------------------------------------------------------------------------

__device__ inline v16bf pack16(v8bf lo, v8bf hi) {
  v16bf r;
#pragma unroll
  for (int i = 0; i < 8; ++i) { r[i] = lo[i]; r[i + 8] = hi[i]; }
  return r;
}

__device__ inline v8f wmma_bf16(v16bf a, v16bf b, v8f c) {
  // D = A(16x32 bf16) * B(32x16 bf16) + C(16x16 f32)
  return __builtin_amdgcn_wmma_f32_16x16x32_bf16(
      /*neg_a=*/false, a, /*neg_b=*/false, b,
      /*c_mod=*/(short)0, c, /*reuse_a=*/false, /*reuse_b=*/false);
}

// A fragment (16x32, row-major source). lane row = lane&15,
// elements 0..7 -> K = kb+e, elements 8..15 -> K = 16+kb+(e-8), kb=(lane>>4)*8
__device__ inline v16bf load_a_frag(const __bf16* __restrict__ A, int lda,
                                    int row, int kk, int lane) {
  int kb = (lane >> 4) << 3;
  const v8bf* p0 = (const v8bf*)(A + (size_t)row * lda + kk + kb);
  const v8bf* p1 = (const v8bf*)(A + (size_t)row * lda + kk + 16 + kb);
  return pack16(*p0, *p1);
}

// B fragment (32x16) from TRANSPOSED source Bt[N][ldb] (row = output column,
// contiguous K). lane col = lane&15, elements 0..7 -> K = kb+e, 8..15 -> K = 16+kb+(e-8), kb=(lane>>4)*8.
__device__ inline v16bf load_b_frag(const __bf16* __restrict__ Bt, int ldb,
                                    int col, int kk, int lane) {
  int kb = (lane >> 4) << 3;
  const v8bf* p0 = (const v8bf*)(Bt + (size_t)col * ldb + kk + kb);
  const v8bf* p1 = (const v8bf*)(Bt + (size_t)col * ldb + kk + 16 + kb);
  return pack16(*p0, *p1);
}

// ---------------------------------------------------------------------------
// conversion / transpose pre-passes
// ---------------------------------------------------------------------------

__global__ void cvt_f32_to_bf16(const float* __restrict__ in,
                                __bf16* __restrict__ out, int n) {
  int i = blockIdx.x * blockDim.x + threadIdx.x;
  if (i < n) out[i] = (__bf16)in[i];
}

// in[R][C] (f32) -> out[C][R] (bf16)
__global__ void transpose_f32_to_bf16(const float* __restrict__ in,
                                      __bf16* __restrict__ out, int R, int C) {
  int i = blockIdx.x * blockDim.x + threadIdx.x;
  if (i < R * C) {
    int r = i / C, c = i % C;
    out[(size_t)c * R + r] = (__bf16)in[i];
  }
}

// ---------------------------------------------------------------------------
// QKV GEMM: [8192,1024]bf16 @ wqt[N=3072][K=1024]bf16 -> scatter into
//   Qh[B,H,T,64], Kh[B,H,T,64], Vt[B,H,64,T]  (all bf16)
// One wave computes a 64(M) x 64(N) tile: 16 accumulators, 16 WMMAs / K-step.
// ---------------------------------------------------------------------------

__global__ __launch_bounds__(128) void qkv_gemm_kernel(
    const __bf16* __restrict__ xb, const __bf16* __restrict__ wqt,
    __bf16* __restrict__ Qh, __bf16* __restrict__ Kh, __bf16* __restrict__ Vt) {
  int lane = threadIdx.x & 31;
  int wid  = blockIdx.x * (blockDim.x >> 5) + (threadIdx.x >> 5);
  const int tilesN = (3 * D_MODEL) / 64;  // 48
  int m0 = (wid / tilesN) * 64;
  int n0 = (wid % tilesN) * 64;
  int ln = lane & 15, lg = lane >> 4;

  v8f acc[4][4];
#pragma unroll
  for (int mi = 0; mi < 4; ++mi)
#pragma unroll
    for (int ni = 0; ni < 4; ++ni) acc[mi][ni] = {};

  for (int kk = 0; kk < D_MODEL; kk += 32) {
    // gfx1250 speculative prefetch for a future K step (uniform guard).
    if (kk + 128 < D_MODEL) {
      __builtin_prefetch(xb + (size_t)(m0 + lane) * D_MODEL + kk + 128, 0, 1);
      __builtin_prefetch(wqt + (size_t)(n0 + lane) * D_MODEL + kk + 128, 0, 1);
    }
    v16bf af[4], bfr[4];
#pragma unroll
    for (int mi = 0; mi < 4; ++mi)
      af[mi] = load_a_frag(xb, D_MODEL, m0 + mi * 16 + ln, kk, lane);
#pragma unroll
    for (int ni = 0; ni < 4; ++ni)
      bfr[ni] = load_b_frag(wqt, D_MODEL, n0 + ni * 16 + ln, kk, lane);
#pragma unroll
    for (int mi = 0; mi < 4; ++mi)
#pragma unroll
      for (int ni = 0; ni < 4; ++ni)
        acc[mi][ni] = wmma_bf16(af[mi], bfr[ni], acc[mi][ni]);
  }

  // epilogue: scatter into head-major layouts
#pragma unroll
  for (int mi = 0; mi < 4; ++mi) {
#pragma unroll
    for (int ni = 0; ni < 4; ++ni) {
      int nBase = n0 + ni * 16;
      int n = nBase + ln;
#pragma unroll
      for (int c = 0; c < 8; ++c) {
        int m = m0 + mi * 16 + lg * 8 + c;
        int b = m >> 11, t = m & (SEQ - 1);
        float v = acc[mi][ni][c];
        if (nBase < D_MODEL) {                       // Q
          int h = n >> 6, d = n & 63;
          Qh[((size_t)((b * N_HEADS + h) * SEQ + t) << 6) + d] = (__bf16)v;
        } else if (nBase < 2 * D_MODEL) {            // K
          int n2 = n - D_MODEL;
          int h = n2 >> 6, d = n2 & 63;
          Kh[((size_t)((b * N_HEADS + h) * SEQ + t) << 6) + d] = (__bf16)v;
        } else {                                     // V (transposed: [64][T])
          int n2 = n - 2 * D_MODEL;
          int h = n2 >> 6, d = n2 & 63;
          Vt[(size_t)((b * N_HEADS + h) * 64 + d) * SEQ + t] = (__bf16)v;
        }
      }
    }
  }
}

// ---------------------------------------------------------------------------
// Flash attention: one wave per (b,h, 32-row q tile) -> 2 row-fragments (QI=2)
// share each K/V fragment (2x arithmetic intensity vs 16-row tiles).
// k processed in 32-wide chunks: S = Q K^T (4 WMMAs per row-frag), online
// softmax in f32 VALU + shfl reductions, P staged C-layout -> A-layout
// through per-wave LDS patch, O += P V (4 WMMAs per row-frag).
// ---------------------------------------------------------------------------

#define QI 2  // 16-row fragments per wave (32 q rows)

__global__ __launch_bounds__(128) void attn_kernel(
    const __bf16* __restrict__ Qh, const __bf16* __restrict__ Kh,
    const __bf16* __restrict__ Vt, __bf16* __restrict__ ab) {
  __shared__ __align__(16) __bf16 Pst[4][QI][16][32];  // 4 waves * 2KB

  int lane  = threadIdx.x & 31;
  int wslot = threadIdx.x >> 5;
  int wid   = blockIdx.x * 4 + wslot;
  int qt = wid & (SEQ / (16 * QI) - 1);  // 0..63
  int bh = wid >> 6;                     // 0..63  (= b*16 + h)
  int q0 = qt * 16 * QI;
  int ln = lane & 15, lg = lane >> 4;

  const __bf16* Qp = Qh + (size_t)bh * SEQ * HEAD_DIM;
  const __bf16* Kp = Kh + (size_t)bh * SEQ * HEAD_DIM;
  const __bf16* Vp = Vt + (size_t)bh * HEAD_DIM * SEQ;

  // Q fragments (two 32-wide d windows per row-fragment), kept in registers.
  v16bf qf[QI][2];
#pragma unroll
  for (int qi = 0; qi < QI; ++qi) {
    qf[qi][0] = load_a_frag(Qp, HEAD_DIM, q0 + qi * 16 + ln, 0, lane);
    qf[qi][1] = load_a_frag(Qp, HEAD_DIM, q0 + qi * 16 + ln, 32, lane);
  }

  v8f o[QI][4];
  float mrow[QI][8], lrow[QI][8];
#pragma unroll
  for (int qi = 0; qi < QI; ++qi) {
#pragma unroll
    for (int dt = 0; dt < 4; ++dt) o[qi][dt] = {};
#pragma unroll
    for (int c = 0; c < 8; ++c) {
      mrow[qi][c] = -__builtin_inff();
      lrow[qi][c] = 0.0f;
    }
  }

  const float scale = 0.125f;  // 1/sqrt(64)
  int jmax = (q0 + 16 * QI - 1) >> 5;

  for (int j = 0; j <= jmax; ++j) {
    int kc0 = j * 32;

    // Prefetch next chunk's K rows / V stripe (uniform guard, gfx1250 path).
    if (j < jmax) {
      __builtin_prefetch(Kp + (size_t)(kc0 + 32 + lane) * HEAD_DIM, 0, 1);
      __builtin_prefetch(Vp + (size_t)(lane * 2) * SEQ + kc0 + 32, 0, 1);
    }

    // ---- shared K fragments: [ktile 0/16] x [d-window 0/32]
    v16bf kfr[4];
    kfr[0] = load_b_frag(Kp, HEAD_DIM, kc0 + ln, 0, lane);
    kfr[1] = load_b_frag(Kp, HEAD_DIM, kc0 + ln, 32, lane);
    kfr[2] = load_b_frag(Kp, HEAD_DIM, kc0 + 16 + ln, 0, lane);
    kfr[3] = load_b_frag(Kp, HEAD_DIM, kc0 + 16 + ln, 32, lane);

    // ---- shared V fragments (issued early, consumed after softmax)
    v16bf vfr[4];
#pragma unroll
    for (int dt = 0; dt < 4; ++dt)
      vfr[dt] = load_b_frag(Vp, SEQ, dt * 16 + ln, kc0, lane);

#pragma unroll
    for (int qi = 0; qi < QI; ++qi) {
      // S = Q K^T for 16x32 chunk (two 16x16 tiles, Kdim = d = 64)
      v8f s0 = {}, s1 = {};
      s0 = wmma_bf16(qf[qi][0], kfr[0], s0);
      s0 = wmma_bf16(qf[qi][1], kfr[1], s0);
      s1 = wmma_bf16(qf[qi][0], kfr[2], s1);
      s1 = wmma_bf16(qf[qi][1], kfr[3], s1);

      // scale, causal mask, online softmax
      int kcol0 = kc0 + ln, kcol1 = kc0 + 16 + ln;
      float p0[8], p1[8], alpha[8];
#pragma unroll
      for (int c = 0; c < 8; ++c) {
        int qrow = q0 + qi * 16 + lg * 8 + c;
        float a0 = s0[c] * scale; if (kcol0 > qrow) a0 = -__builtin_inff();
        float a1 = s1[c] * scale; if (kcol1 > qrow) a1 = -__builtin_inff();
        float mx = fmaxf(a0, a1);
#pragma unroll
        for (int off = 1; off < 16; off <<= 1)
          mx = fmaxf(mx, __shfl_xor(mx, off, 16));
        float mnew = fmaxf(mrow[qi][c], mx);
        float e0 = __expf(a0 - mnew);
        float e1 = __expf(a1 - mnew);
        float sum = e0 + e1;
#pragma unroll
        for (int off = 1; off < 16; off <<= 1)
          sum += __shfl_xor(sum, off, 16);
        alpha[c] = __expf(mrow[qi][c] - mnew);
        lrow[qi][c] = lrow[qi][c] * alpha[c] + sum;
        mrow[qi][c] = mnew;
        p0[c] = e0; p1[c] = e1;
      }

      // rescale O accumulators by alpha (row-matched element layout)
#pragma unroll
      for (int dt = 0; dt < 4; ++dt)
#pragma unroll
        for (int c = 0; c < 8; ++c) o[qi][dt][c] *= alpha[c];

      // P: C-fragment layout -> A-fragment layout via LDS patch
#pragma unroll
      for (int c = 0; c < 8; ++c) {
        int r = lg * 8 + c;
        Pst[wslot][qi][r][ln]      = (__bf16)p0[c];
        Pst[wslot][qi][r][16 + ln] = (__bf16)p1[c];
      }
    }

    asm volatile("s_wait_dscnt 0" ::: "memory");

#pragma unroll
    for (int qi = 0; qi < QI; ++qi) {
      v16bf pf;
      {
        int kb8 = lg * 8;
        const v8bf* pp0 = (const v8bf*)&Pst[wslot][qi][ln][kb8];
        const v8bf* pp1 = (const v8bf*)&Pst[wslot][qi][ln][16 + kb8];
        pf = pack16(*pp0, *pp1);
      }
      // O += P V  (V stored transposed: row d, contiguous k)
#pragma unroll
      for (int dt = 0; dt < 4; ++dt)
        o[qi][dt] = wmma_bf16(pf, vfr[dt], o[qi][dt]);
    }
  }

  // ---- normalize and store to [B,T,C] bf16 (head-interleaved)
  int b = bh >> 4, h = bh & 15;
#pragma unroll
  for (int qi = 0; qi < QI; ++qi) {
#pragma unroll
    for (int dt = 0; dt < 4; ++dt) {
#pragma unroll
      for (int c = 0; c < 8; ++c) {
        int t = q0 + qi * 16 + lg * 8 + c;
        int d = dt * 16 + ln;
        float val = o[qi][dt][c] / lrow[qi][c];
        ab[(size_t)(b * SEQ + t) * D_MODEL + h * HEAD_DIM + d] = (__bf16)val;
      }
    }
  }
}

// ---------------------------------------------------------------------------
// Output projection: out = ab[8192,1024] @ wpt^T + b_proj  (fp32 out)
// ---------------------------------------------------------------------------

__global__ __launch_bounds__(128) void proj_gemm_kernel(
    const __bf16* __restrict__ ab, const __bf16* __restrict__ wpt,
    const float* __restrict__ bp, float* __restrict__ out) {
  int lane = threadIdx.x & 31;
  int wid  = blockIdx.x * (blockDim.x >> 5) + (threadIdx.x >> 5);
  const int tilesN = D_MODEL / 64;  // 16
  int m0 = (wid / tilesN) * 64;
  int n0 = (wid % tilesN) * 64;
  int ln = lane & 15, lg = lane >> 4;

  v8f acc[4][4];
#pragma unroll
  for (int mi = 0; mi < 4; ++mi)
#pragma unroll
    for (int ni = 0; ni < 4; ++ni) acc[mi][ni] = {};

  for (int kk = 0; kk < D_MODEL; kk += 32) {
    if (kk + 128 < D_MODEL) {
      __builtin_prefetch(ab + (size_t)(m0 + lane) * D_MODEL + kk + 128, 0, 1);
      __builtin_prefetch(wpt + (size_t)(n0 + lane) * D_MODEL + kk + 128, 0, 1);
    }
    v16bf af[4], bfr[4];
#pragma unroll
    for (int mi = 0; mi < 4; ++mi)
      af[mi] = load_a_frag(ab, D_MODEL, m0 + mi * 16 + ln, kk, lane);
#pragma unroll
    for (int ni = 0; ni < 4; ++ni)
      bfr[ni] = load_b_frag(wpt, D_MODEL, n0 + ni * 16 + ln, kk, lane);
#pragma unroll
    for (int mi = 0; mi < 4; ++mi)
#pragma unroll
      for (int ni = 0; ni < 4; ++ni)
        acc[mi][ni] = wmma_bf16(af[mi], bfr[ni], acc[mi][ni]);
  }

#pragma unroll
  for (int mi = 0; mi < 4; ++mi) {
#pragma unroll
    for (int ni = 0; ni < 4; ++ni) {
      int n = n0 + ni * 16 + ln;
      float bias = bp[n];
#pragma unroll
      for (int c = 0; c < 8; ++c) {
        int m = m0 + mi * 16 + lg * 8 + c;
        out[(size_t)m * D_MODEL + n] = acc[mi][ni][c] + bias;
      }
    }
  }
}

// ---------------------------------------------------------------------------
// launch
// ---------------------------------------------------------------------------

extern "C" void kernel_launch(void* const* d_in, const int* in_sizes, int n_in,
                              void* d_out, int out_size, void* d_ws, size_t ws_size,
                              hipStream_t stream) {
  const float* x     = (const float*)d_in[0];  // [4,2048,1024]
  const float* wqkv  = (const float*)d_in[1];  // [1024,3072]
  const float* wproj = (const float*)d_in[2];  // [1024,1024]
  const float* bproj = (const float*)d_in[3];  // [1024]
  float* out = (float*)d_out;

  // Workspace layout (bf16 elements). Total = 88 MB.
  __bf16* ws = (__bf16*)d_ws;
  size_t off = 0;
  __bf16* xb  = ws + off; off += (size_t)M_TOT * D_MODEL;          // 8192*1024
  __bf16* wqt = ws + off; off += (size_t)3 * D_MODEL * D_MODEL;    // 3072*1024
  __bf16* wpt = ws + off; off += (size_t)D_MODEL * D_MODEL;        // 1024*1024
  __bf16* Qh  = ws + off; off += (size_t)M_TOT * D_MODEL;          // [B,H,T,64]
  __bf16* Kh  = ws + off; off += (size_t)M_TOT * D_MODEL;          // [B,H,T,64]
  __bf16* Vt  = ws + off; off += (size_t)M_TOT * D_MODEL;          // [B,H,64,T]
  __bf16* ab  = ws + off; off += (size_t)M_TOT * D_MODEL;          // [B,T,C]

  // 1) bf16 conversions / weight transposes
  {
    int n = M_TOT * D_MODEL;  // 8388608
    cvt_f32_to_bf16<<<n / 256, 256, 0, stream>>>(x, xb, n);
  }
  {
    int n = D_MODEL * 3 * D_MODEL;  // 3145728
    transpose_f32_to_bf16<<<n / 256, 256, 0, stream>>>(wqkv, wqt, D_MODEL, 3 * D_MODEL);
  }
  {
    int n = D_MODEL * D_MODEL;  // 1048576
    transpose_f32_to_bf16<<<n / 256, 256, 0, stream>>>(wproj, wpt, D_MODEL, D_MODEL);
  }

  // 2) QKV GEMM: (8192/64)*(3072/64) = 6144 waves, 4 waves/block
  qkv_gemm_kernel<<<6144 / 4, 128, 0, stream>>>(xb, wqt, Qh, Kh, Vt);

  // 3) attention: B*H*(T/32) = 4096 waves, 4 waves/block
  attn_kernel<<<4096 / 4, 128, 0, stream>>>(Qh, Kh, Vt, ab);

  // 4) projection: (8192/64)*(1024/64) = 2048 waves, 4 waves/block
  proj_gemm_kernel<<<2048 / 4, 128, 0, stream>>>(ab, wpt, bproj, out);
}
